// OptimizedMoELayer_57655640982120
// MI455X (gfx1250) — compile-verified, matrix operand order checked by reading, not run on hardware
//
#include <hip/hip_runtime.h>
#include <cstdint>
#include <cstddef>

// MoE top-2 FFN for MI455X (gfx1250, wave32, WMMA bf16).
// T=8192 tokens, D=1024, H=4096, E=8 experts, top-2 routing.
// BM=64 token tile held in 128KB LDS (CDNA5: 320KB/WGP) to cut L2 weight traffic 4x vs BM=16.

#define T_TOK 8192
#define DIM   1024
#define HID   4096
#define NEXP  8
#define BM    64     // token tile (M): 4 WMMA M-tiles
#define HB    128    // hidden chunk per outer iteration

typedef __attribute__((ext_vector_type(16))) __bf16 v16bf;
typedef __attribute__((ext_vector_type(8)))  float  v8f;

union BfFrag { uint4 q[2]; v16bf v; };   // 32 bytes = 16 bf16 per lane

__device__ __forceinline__ unsigned short f2bf(float f) {
  union { float f; unsigned int u; } c; c.f = f;
  unsigned int r = c.u + 0x7FFFu + ((c.u >> 16) & 1u);  // round-to-nearest-even
  return (unsigned short)(r >> 16);
}

// ---------------- init: zero output + counters ----------------
__global__ void zero_kernel(float* out, size_t n, int* cnt, float* psum) {
  size_t i = (size_t)blockIdx.x * blockDim.x + threadIdx.x;
  size_t stride = (size_t)gridDim.x * blockDim.x;
  for (; i < n; i += stride) out[i] = 0.0f;
  if (blockIdx.x == 0 && threadIdx.x < NEXP) {
    cnt[threadIdx.x] = 0;
    psum[threadIdx.x] = 0.0f;
  }
}

// ---------------- f32 -> bf16 plain convert (for x) ----------------
__global__ void conv_kernel(const float* __restrict__ in, unsigned short* __restrict__ o, size_t n) {
  size_t i = (size_t)blockIdx.x * blockDim.x + threadIdx.x;
  size_t stride = (size_t)gridDim.x * blockDim.x;
  for (; i < n; i += stride) o[i] = f2bf(in[i]);
}

// ------- f32 [E][R][C] row-major -> bf16 [E][C][R] (N-major for WMMA B frags) -------
__global__ __launch_bounds__(256) void transconv_kernel(const float* __restrict__ in,
                                                        unsigned short* __restrict__ o,
                                                        int R, int C) {
  __shared__ float tile[32][33];
  size_t base = (size_t)blockIdx.z * (size_t)R * (size_t)C;
  int c0 = blockIdx.x * 32, r0 = blockIdx.y * 32;
  for (int i = threadIdx.y; i < 32; i += 8)
    tile[i][threadIdx.x] = in[base + (size_t)(r0 + i) * C + c0 + threadIdx.x];
  __syncthreads();
  for (int i = threadIdx.y; i < 32; i += 8)
    o[base + (size_t)(c0 + i) * R + r0 + threadIdx.x] = f2bf(tile[threadIdx.x][i]);
}

// ---------------- gating: logits, softmax, top-2, routing lists ----------------
__global__ __launch_bounds__(256) void gate_kernel(const float* __restrict__ x,
                                                   const float* __restrict__ gw,
                                                   float* __restrict__ comb,
                                                   int* __restrict__ tokens,
                                                   int* __restrict__ cnt,
                                                   float* __restrict__ psum) {
  __shared__ float wprob[8][NEXP];
  const int lane = threadIdx.x & 31;
  const int wv = threadIdx.x >> 5;
  const int t = blockIdx.x * 8 + wv;

  float acc[NEXP];
#pragma unroll
  for (int e = 0; e < NEXP; e++) acc[e] = 0.0f;

  const float* xr = x + (size_t)t * DIM;
  for (int d = lane; d < DIM; d += 32) {
    float xv = xr[d];
    const float* g = gw + d * NEXP;
#pragma unroll
    for (int e = 0; e < NEXP; e++) acc[e] += xv * g[e];
  }
#pragma unroll
  for (int e = 0; e < NEXP; e++) {
    for (int off = 16; off > 0; off >>= 1) acc[e] += __shfl_xor(acc[e], off, 32);
  }

  if (lane == 0) {
    // full softmax (for load-balance stats)
    float mx = acc[0];
#pragma unroll
    for (int e = 1; e < NEXP; e++) mx = fmaxf(mx, acc[e]);
    float p[NEXP], s = 0.0f;
#pragma unroll
    for (int e = 0; e < NEXP; e++) { p[e] = __expf(acc[e] - mx); s += p[e]; }
    float inv = 1.0f / s;
#pragma unroll
    for (int e = 0; e < NEXP; e++) wprob[wv][e] = p[e] * inv;

    // top-2
    int i1 = 0;
#pragma unroll
    for (int e = 1; e < NEXP; e++) if (acc[e] > acc[i1]) i1 = e;
    int i2 = (i1 == 0) ? 1 : 0;
#pragma unroll
    for (int e = 0; e < NEXP; e++) if (e != i1 && acc[e] > acc[i2]) i2 = e;
    float m2 = fmaxf(acc[i1], acc[i2]);
    float e1 = __expf(acc[i1] - m2), e2 = __expf(acc[i2] - m2);
    float is = 1.0f / (e1 + e2);
    float cw[NEXP];
#pragma unroll
    for (int e = 0; e < NEXP; e++) cw[e] = 0.0f;
    cw[i1] = e1 * is;
    cw[i2] = e2 * is;
#pragma unroll
    for (int e = 0; e < NEXP; e++) comb[(size_t)t * NEXP + e] = cw[e];

    int p1 = atomicAdd(&cnt[i1], 1);
    tokens[i1 * T_TOK + p1] = t;
    int p2 = atomicAdd(&cnt[i2], 1);
    tokens[i2 * T_TOK + p2] = t;
  }
  __syncthreads();
  if (threadIdx.x < NEXP) {
    float s = 0.0f;
#pragma unroll
    for (int w = 0; w < 8; w++) s += wprob[w][threadIdx.x];
    __hip_atomic_fetch_add(&psum[threadIdx.x], s, __ATOMIC_RELAXED, __HIP_MEMORY_SCOPE_AGENT);
  }
}

// ---------------- fused expert FFN: silu(x Wg) * (x Wu) @ Wd, scatter-add ----------------
// 512 threads = 16 waves. BM=64 tokens per block.
// Phase 1: wave wv computes h columns [16*(wv&7), +16) of the HB chunk for
//          M-tile pair {2*(wv>>3), 2*(wv>>3)+1}.
// Phase 2: wave wv owns output columns [wv*64, wv*64+64): 4x4 WMMA cross product.
__global__ __launch_bounds__(512) void ffn_kernel(const unsigned short* __restrict__ xb,
                                                  const unsigned short* __restrict__ wgT,
                                                  const unsigned short* __restrict__ wuT,
                                                  const unsigned short* __restrict__ wdT,
                                                  const float* __restrict__ comb,
                                                  const int* __restrict__ tokens,
                                                  const int* __restrict__ cnt,
                                                  float* __restrict__ out) {
  const int e = blockIdx.y;
  const int count = cnt[e];
  if ((int)blockIdx.x * BM >= count) return;

  const int tid = threadIdx.x;
  const int lane = tid & 31;
  const int wv = tid >> 5;            // 0..15

  __shared__ __align__(16) unsigned short xs[BM * DIM];  // 128 KB: 64 token rows, bf16
  __shared__ __align__(16) unsigned short hs[BM * HB];   // 16 KB: 64 x 128 h chunk, bf16
  __shared__ int   stok[BM];
  __shared__ float swgt[BM];

  if (tid < BM) {
    int r = blockIdx.x * BM + tid;
    int tk = (r < count) ? tokens[e * T_TOK + r] : 0;
    stok[tid] = tk;
    swgt[tid] = (r < count) ? comb[(size_t)tk * NEXP + e] : 0.0f;
  }
  __syncthreads();

  // stage 64 token rows of bf16 x into LDS (each row 2 KB = 128 uint4)
  for (int i = tid; i < BM * 128; i += 512) {
    int m = i >> 7, c = i & 127;
    ((uint4*)xs)[m * 128 + c] = *((const uint4*)(xb + (size_t)stok[m] * DIM) + c);
  }
  __syncthreads();

  const unsigned short* wgE = wgT + (size_t)e * HID * DIM;  // [H][D] N-major
  const unsigned short* wuE = wuT + (size_t)e * HID * DIM;  // [H][D]
  const unsigned short* wdE = wdT + (size_t)e * DIM * HID;  // [D][H] N-major

  v8f yacc[4][4];                       // [m-tile][n-tile] C fragments
#pragma unroll
  for (int i = 0; i < 4; i++)
#pragma unroll
    for (int j = 0; j < 4; j++) yacc[i][j] = (v8f){};

  const int mA = lane & 15;             // A-fragment row within a 16-row tile
  const int khalf = (lane >> 4) * 16;   // which half of the K=32 slab this lane holds
  const int mC = 8 * (lane >> 4);       // C-layout: VGPR v -> M = v + mC
  const int nC = lane & 15;

  const int nt1 = wv & 7;               // phase-1 column tile
  const int mp = (wv >> 3) * 2;         // phase-1 first M-tile of the pair

  for (int hb = 0; hb < HID; hb += HB) {
    // ---- phase 1: h[64, HB] = silu(x Wg) * (x Wu)
    v8f g0 = (v8f){}, u0 = (v8f){}, g1 = (v8f){}, u1 = (v8f){};
    const int n1 = hb + nt1 * 16 + (lane & 15);
    const unsigned short* wg_row = wgE + (size_t)n1 * DIM + khalf;
    const unsigned short* wu_row = wuE + (size_t)n1 * DIM + khalf;
    const unsigned short* xrow0 = xs + (mp * 16 + mA) * DIM + khalf;
    const unsigned short* xrow1 = xs + (mp * 16 + 16 + mA) * DIM + khalf;
    for (int k = 0; k < DIM; k += 32) {
      BfFrag a0, a1, bg, bu;
      a0.q[0] = *(const uint4*)(xrow0 + k);
      a0.q[1] = *(const uint4*)(xrow0 + k + 8);
      a1.q[0] = *(const uint4*)(xrow1 + k);
      a1.q[1] = *(const uint4*)(xrow1 + k + 8);
      bg.q[0] = *(const uint4*)(wg_row + k);
      bg.q[1] = *(const uint4*)(wg_row + k + 8);
      bu.q[0] = *(const uint4*)(wu_row + k);
      bu.q[1] = *(const uint4*)(wu_row + k + 8);
      g0 = __builtin_amdgcn_wmma_f32_16x16x32_bf16(false, a0.v, false, bg.v, (short)0, g0, false, false);
      u0 = __builtin_amdgcn_wmma_f32_16x16x32_bf16(false, a0.v, false, bu.v, (short)0, u0, false, false);
      g1 = __builtin_amdgcn_wmma_f32_16x16x32_bf16(false, a1.v, false, bg.v, (short)0, g1, false, false);
      u1 = __builtin_amdgcn_wmma_f32_16x16x32_bf16(false, a1.v, false, bu.v, (short)0, u1, false, false);
    }
    __syncthreads();   // previous iteration's phase-2 reads of hs are done
#pragma unroll
    for (int v = 0; v < 8; v++) {
      float gg0 = g0[v];
      float h0 = (gg0 / (1.0f + __expf(-gg0))) * u0[v];
      hs[(mp * 16 + mC + v) * HB + nt1 * 16 + nC] = f2bf(h0);
      float gg1 = g1[v];
      float h1 = (gg1 / (1.0f + __expf(-gg1))) * u1[v];
      hs[(mp * 16 + 16 + mC + v) * HB + nt1 * 16 + nC] = f2bf(h1);
    }
    __syncthreads();   // hs fully written

    // ---- phase 2: y[64, wv*64 .. +64) += h[64, HB] @ Wd[hb:hb+HB, :]
    for (int kk = 0; kk < HB; kk += 32) {
      BfFrag a[4];
#pragma unroll
      for (int mt = 0; mt < 4; mt++) {
        a[mt].q[0] = *(const uint4*)(hs + (mt * 16 + mA) * HB + kk + khalf);
        a[mt].q[1] = *(const uint4*)(hs + (mt * 16 + mA) * HB + kk + khalf + 8);
      }
#pragma unroll
      for (int nt = 0; nt < 4; nt++) {
        const int n2 = wv * 64 + nt * 16 + (lane & 15);
        const unsigned short* wd_row = wdE + (size_t)n2 * HID + hb + kk + khalf;
        BfFrag b2;
        b2.q[0] = *(const uint4*)(wd_row);
        b2.q[1] = *(const uint4*)(wd_row + 8);
#pragma unroll
        for (int mt = 0; mt < 4; mt++) {
          yacc[mt][nt] = __builtin_amdgcn_wmma_f32_16x16x32_bf16(false, a[mt].v, false, b2.v,
                                                                 (short)0, yacc[mt][nt], false, false);
        }
      }
    }
  }

  // ---- scatter with combine weight (a token can belong to 2 experts -> atomic add)
#pragma unroll
  for (int mt = 0; mt < 4; mt++) {
#pragma unroll
    for (int nt = 0; nt < 4; nt++) {
#pragma unroll
      for (int v = 0; v < 8; v++) {
        const int m = mt * 16 + mC + v;
        const int n = wv * 64 + nt * 16 + nC;
        float val = yacc[mt][nt][v] * swgt[m];
        __hip_atomic_fetch_add(out + (size_t)stok[m] * DIM + n, val,
                               __ATOMIC_RELAXED, __HIP_MEMORY_SCOPE_AGENT);
      }
    }
  }
}

// ---------------- finalize: mean_probs + load balance loss ----------------
__global__ void finalize_kernel(const float* __restrict__ psum, float* __restrict__ dst) {
  if (threadIdx.x == 0) {
    float loss = 0.0f;
    for (int e = 0; e < NEXP; e++) {
      float m = psum[e] / (float)T_TOK;
      dst[1 + e] = m;
      loss += m * m;
    }
    dst[0] = (float)NEXP * loss;
  }
}

// ---------------- host ----------------
extern "C" void kernel_launch(void* const* d_in, const int* in_sizes, int n_in,
                              void* d_out, int out_size, void* d_ws, size_t ws_size,
                              hipStream_t stream) {
  const float* x      = (const float*)d_in[0];   // [T, D]
  const float* gate_w = (const float*)d_in[1];   // [D, E]
  const float* w_gate = (const float*)d_in[2];   // [E, D, H]
  const float* w_up   = (const float*)d_in[3];   // [E, D, H]
  const float* w_down = (const float*)d_in[4];   // [E, H, D]
  float* out = (float*)d_out;                    // [T*D] + loss[1] + mean_probs[8]

  // workspace layout (~208.5 MB total)
  char* ws = (char*)d_ws;
  const size_t SZ_XB  = (size_t)T_TOK * DIM * 2;        // 16 MB
  const size_t SZ_W   = (size_t)NEXP * DIM * HID * 2;   // 64 MB each
  unsigned short* xb  = (unsigned short*)(ws);
  unsigned short* wgT = (unsigned short*)(ws + SZ_XB);
  unsigned short* wuT = (unsigned short*)(ws + SZ_XB + SZ_W);
  unsigned short* wdT = (unsigned short*)(ws + SZ_XB + 2 * SZ_W);
  float* comb   = (float*)(ws + SZ_XB + 3 * SZ_W);
  int*   tokens = (int*)(ws + SZ_XB + 3 * SZ_W + (size_t)T_TOK * NEXP * 4);
  int*   cnt    = (int*)(ws + SZ_XB + 3 * SZ_W + 2 * (size_t)T_TOK * NEXP * 4);
  float* psum   = (float*)((char*)cnt + 64);

  const size_t n_out = (size_t)T_TOK * DIM + 1 + NEXP;

  zero_kernel<<<4096, 256, 0, stream>>>(out, n_out, cnt, psum);
  conv_kernel<<<4096, 256, 0, stream>>>(x, xb, (size_t)T_TOK * DIM);
  // w_gate/w_up: [E][D=1024][H=4096] -> [E][H][D] bf16
  transconv_kernel<<<dim3(HID / 32, DIM / 32, NEXP), dim3(32, 8), 0, stream>>>(w_gate, wgT, DIM, HID);
  transconv_kernel<<<dim3(HID / 32, DIM / 32, NEXP), dim3(32, 8), 0, stream>>>(w_up, wuT, DIM, HID);
  // w_down: [E][H=4096][D=1024] -> [E][D][H] bf16
  transconv_kernel<<<dim3(DIM / 32, HID / 32, NEXP), dim3(32, 8), 0, stream>>>(w_down, wdT, HID, DIM);

  gate_kernel<<<T_TOK / 8, 256, 0, stream>>>(x, gate_w, comb, tokens, cnt, psum);

  // grid: x = token blocks (worst case all tokens on one expert), y = expert
  ffn_kernel<<<dim3(T_TOK / BM, NEXP), 512, 0, stream>>>(xb, wgT, wuT, wdT,
                                                         comb, tokens, cnt, out);

  finalize_kernel<<<1, 32, 0, stream>>>(psum, out + (size_t)T_TOK * DIM);
}
